// StareEncoder_20950850470387
// MI455X (gfx1250) — matrix operand order repaired and checked by reading, not compiled, not run on hardware
//
#include <hip/hip_runtime.h>
#include <hip/hip_fp16.h>

// ---------------- problem constants (match reference) ----------------
constexpr int kEnt   = 100000;   // NUM_ENT
constexpr int kRels  = 1000;     // 2*NUM_REL
constexpr int kD     = 200;      // embedding dim
constexpr int kDH    = 100;      // D/2
constexpr int kDP    = 224;      // D padded to 7*32 for WMMA K
constexpr int kDPH   = 112;      // kDP/2 (pairs)
constexpr int kNT    = 13;       // N tiles of 16 (208 >= 200)
constexpr int kKC    = 7;        // K chunks of 32 (224)
constexpr int kE     = 600000;   // edges
constexpr int kHalfE = 300000;
constexpr int kNQ    = 200000;   // qualifiers
constexpr int kHQ    = 100000;
constexpr int kBS    = 256;

typedef __attribute__((ext_vector_type(16))) _Float16  v16h;
typedef __attribute__((ext_vector_type(8)))  float     v8f;
typedef __attribute__((ext_vector_type(8)))  unsigned  v8u;
typedef __attribute__((ext_vector_type(4)))  unsigned  v4u;

// Packed f32x2 -> f16x2 via hardware v_cvt_pk_rtz_f16_f32 (single VALU op).
static __device__ inline unsigned to_f16x2(float lo, float hi) {
  auto r = __builtin_amdgcn_cvt_pkrtz(lo, hi);
  return __builtin_bit_cast(unsigned, r);
}

static __device__ inline v8f wmma_f16(v16h a, v16h b, v8f c) {
  // (neg_a, A, neg_b, B, c_mod, C, reuse_a, reuse_b)
  return __builtin_amdgcn_wmma_f32_16x16x32_f16(false, a, false, b, (short)0, c,
                                                false, false);
}

// B fragment from pre-swizzled weight buffer: 16 contiguous f16 per lane.
static __device__ inline v16h load_bfrag(const _Float16* __restrict__ F, int lane,
                                         int kc, int nt) {
  return *(const v16h*)(F + ((((kc * kNT) + nt) * 32 + lane) << 4));
}

// Scatter a packed f16 pair (even k) into an LDS tile stored in WMMA-A
// fragment order: element index = ((kc*32 + lsel*16 + m)<<4) + i.
// ISA 16-bit A 16x32: lane = m + 16*((k%32 / 8) & 1), i = (k&7) + ((k%32)>=16 ? 8 : 0).
static __device__ inline void store_pk(unsigned* __restrict__ tileu, int m, int k,
                                       unsigned v) {
  int kc = k >> 5, r = k & 31;
  int lsel = (r >> 3) & 1;
  int i = (r & 7) + ((r >> 4) << 3);
  int ei = ((kc * 32 + lsel * 16 + m) << 4) + i;
  tileu[ei >> 1] = v;
}

// A fragment from f16 fragment-layout LDS tile: one 32-B load.
static __device__ inline v16h afrag_ldsh(const _Float16* __restrict__ tb, int lane,
                                         int kc) {
  return *(const v16h*)(tb + ((kc * 32 + lane) << 4));
}

// ---------------- small prep kernels ----------------

// x -> padded f32 (stride kDP) + padded f16 copy. One thread per element pair.
__global__ void k_init_x(const float* __restrict__ xin, float* __restrict__ xout,
                         unsigned* __restrict__ xhf) {
  long idx = (long)blockIdx.x * blockDim.x + threadIdx.x;
  if (idx >= (long)kEnt * kDPH) return;
  int k2 = 2 * (int)(idx % kDPH);
  long v = idx / kDPH;
  float o0 = 0.0f, o1 = 0.0f;
  if (k2 < kD) {
    o0 = xin[v * kD + k2];
    o1 = xin[v * kD + k2 + 1];
  }
  xout[v * kDP + k2] = o0;
  xout[v * kDP + k2 + 1] = o1;
  xhf[idx] = to_f16x2(o0, o1);
}

// relbuf[0..999] = r, relbuf[1000] = loop_rel  (stride kD)
__global__ void k_prep_rel(const float* __restrict__ rprev,
                           const float* __restrict__ loop_rel,
                           float* __restrict__ relbuf) {
  int idx = blockIdx.x * blockDim.x + threadIdx.x;
  if (idx >= (kRels + 1) * kD) return;
  int k = idx % kD, r = idx / kD;
  relbuf[idx] = (r < kRels) ? rprev[r * kD + k] : loop_rel[k];
}

// Swizzle a [kD x kD] f32 weight (row-major [k][n]) into WMMA-B f16 fragments.
// One thread per packed pair.
__global__ void k_prep_wfrag(const float* __restrict__ W, unsigned* __restrict__ F) {
  int idx = blockIdx.x * blockDim.x + threadIdx.x;
  if (idx >= kKC * kNT * 32 * 8) return;
  int i    = (idx & 7) * 2;
  int lane = (idx >> 3) & 31;
  int rest = idx >> 8;
  int nt = rest % kNT, kc = rest / kNT;
  int n = nt * 16 + (lane & 15);
  int k = kc * 32 + ((lane >> 4) << 3) + (i & 7) + ((i >> 3) << 4);
  float f0 = (k < kD && n < kD) ? W[k * kD + n] : 0.0f;
  float f1 = (k + 1 < kD && n < kD) ? W[(k + 1) * kD + n] : 0.0f;
  F[idx] = to_f16x2(f0, f1);
}

// Per-direction degree (scatter of ones at row).
__global__ void k_degree(const int* __restrict__ ei, float* __restrict__ deg) {
  int e = blockIdx.x * blockDim.x + threadIdx.x;
  if (e >= kE) return;
  atomicAdd(deg + (e < kHalfE ? 0 : kEnt) + ei[e], 1.0f);
}

__global__ void k_dinv(float* __restrict__ deg) {
  int i = blockIdx.x * blockDim.x + threadIdx.x;
  if (i >= 2 * kEnt) return;
  float d = deg[i];
  deg[i] = (d > 0.0f) ? rsqrtf(d) : 0.0f;
}

// Qualifier phi: q_sum[edge] += rotate(x[qent], rel[qrel]); one thread per (q, k<100).
__global__ void k_qual(const int* __restrict__ quals, const float* __restrict__ x,
                       const float* __restrict__ relbuf, float* __restrict__ qsum) {
  long idx = (long)blockIdx.x * blockDim.x + threadIdx.x;
  if (idx >= (long)kNQ * kDH) return;
  int k = (int)(idx % kDH);
  int q = (int)(idx / kDH);
  int qrel = quals[q];
  int qent = quals[kNQ + q];
  int qeid = quals[2 * kNQ + q];
  long erow = (q < kHQ) ? qeid : (long)kHalfE + qeid;
  float xr = x[(long)qent * kDP + k], xi = x[(long)qent * kDP + k + kDH];
  float rr = relbuf[qrel * kD + k],   ri = relbuf[qrel * kD + k + kDH];
  atomicAdd(&qsum[erow * kD + k],       xr * rr - xi * ri);
  atomicAdd(&qsum[erow * kD + k + kDH], xr * ri + xi * rr);
}

// ---------------- WMMA GEMM kernels ----------------

// In-place rows GEMM: A[r,:] = A[r,:] @ W  (A stride kD, M rows). 8 waves/block,
// each wave owns 16 rows; all reads issue before its writes.
__global__ __launch_bounds__(256) void k_gemm_inplace(float* __restrict__ A,
                                                      const _Float16* __restrict__ F,
                                                      long M) {
  int wave = blockIdx.x * (blockDim.x >> 5) + (threadIdx.x >> 5);
  int lane = threadIdx.x & 31;
  long r0 = (long)wave * 16;
  if (r0 >= M) return;  // wave-uniform
  long mrow = r0 + (lane & 15);
  if (mrow >= M) mrow = M - 1;
  const float* rowp = A + mrow * kD;
  int kb0 = (lane >> 4) << 3;
  v8f acc[kNT];
#pragma unroll
  for (int t = 0; t < kNT; ++t)
#pragma unroll
    for (int j = 0; j < 8; ++j) acc[t][j] = 0.0f;
#pragma unroll
  for (int kc = 0; kc < kKC; ++kc) {
    int kb = kc * 32 + kb0;
    v8u au;
#pragma unroll
    for (int jj = 0; jj < 4; ++jj) {
      int k = kb + 2 * jj;
      float f0 = (k < kD) ? rowp[k] : 0.0f;
      float f1 = (k + 1 < kD) ? rowp[k + 1] : 0.0f;
      au[jj] = to_f16x2(f0, f1);
    }
#pragma unroll
    for (int jj = 0; jj < 4; ++jj) {
      int k = kb + 16 + 2 * jj;
      float f0 = (k < kD) ? rowp[k] : 0.0f;
      float f1 = (k + 1 < kD) ? rowp[k + 1] : 0.0f;
      au[4 + jj] = to_f16x2(f0, f1);
    }
    v16h a = __builtin_bit_cast(v16h, au);
#pragma unroll
    for (int t = 0; t < kNT; ++t)
      acc[t] = wmma_f16(a, load_bfrag(F, lane, kc, t), acc[t]);
  }
  int half = lane >> 4, nn = lane & 15;
#pragma unroll
  for (int t = 0; t < kNT; ++t) {
    int n = t * 16 + nn;
    if (n >= kD) continue;
#pragma unroll
    for (int j = 0; j < 8; ++j) {
      long m = r0 + half * 8 + j;
      if (m < M) A[m * kD + n] = acc[t][j];
    }
  }
}

// Fused edge-message kernel: h = rotate(x[col], 0.5*rel[etype] + 0.5*q_agg[e]);
// msg = (h @ W) * norm; atomic scatter-add into agg[row]. 8 waves x 16 edges.
__global__ __launch_bounds__(256) void k_message(
    const int* __restrict__ ei, const int* __restrict__ etype,
    const float* __restrict__ x, const float* __restrict__ relbuf,
    const float* __restrict__ qagg, const float* __restrict__ dinv,
    const _Float16* __restrict__ F, float* __restrict__ agg, int ebase) {
  __shared__ alignas(32) unsigned shtile[8 * kKC * 32 * 8];  // 57344 B (f16 frags)
  __shared__ float shnorm[128];
  __shared__ int shrow[128], shcol[128], shet[128];
  int warp = threadIdx.x >> 5, lane = threadIdx.x & 31;
  unsigned* tileu = shtile + warp * (kKC * 32 * 8);
  const _Float16* tb = (const _Float16*)tileu;
  long e0 = (long)(blockIdx.x * 8 + warp) * 16 + ebase;
  long elim = (long)ebase + kHalfE;
  if (lane < 16) {
    long e = e0 + lane;
    bool valid = e < elim;
    long ec = valid ? e : (elim - 1);
    int row = ei[ec];
    int col = ei[(long)kE + ec];
    shrow[warp * 16 + lane] = row;
    shcol[warp * 16 + lane] = col;
    shet[warp * 16 + lane] = etype[ec];
    shnorm[warp * 16 + lane] = valid ? dinv[row] * dinv[col] : 0.0f;
  }
  __syncthreads();
  // rotate staging: pairs (k2, k2+1) and (k2+100, k2+101), packed to f16.
  for (int t = lane; t < 16 * 50; t += 32) {
    int m = t / 50, k2 = 2 * (t % 50);
    long e = e0 + m;
    long ec = (e < elim) ? e : (elim - 1);
    int col = shcol[warp * 16 + m];
    int et = shet[warp * 16 + m];
    const float* xp = x + (long)col * kDP;
    const float* rp = relbuf + (long)et * kD;
    const float* qp = qagg + ec * kD;
    float xr0 = xp[k2], xr1 = xp[k2 + 1];
    float xi0 = xp[k2 + kDH], xi1 = xp[k2 + kDH + 1];
    float rr0 = 0.5f * rp[k2] + 0.5f * qp[k2];
    float rr1 = 0.5f * rp[k2 + 1] + 0.5f * qp[k2 + 1];
    float ri0 = 0.5f * rp[k2 + kDH] + 0.5f * qp[k2 + kDH];
    float ri1 = 0.5f * rp[k2 + kDH + 1] + 0.5f * qp[k2 + kDH + 1];
    store_pk(tileu, m, k2, to_f16x2(xr0 * rr0 - xi0 * ri0, xr1 * rr1 - xi1 * ri1));
    store_pk(tileu, m, k2 + kDH,
             to_f16x2(xr0 * ri0 + xi0 * rr0, xr1 * ri1 + xi1 * rr1));
  }
  for (int t = lane; t < 16 * 12; t += 32)  // zero pad k = 200..223
    store_pk(tileu, t / 12, kD + 2 * (t % 12), 0u);
  __syncthreads();
  v8f acc[kNT];
#pragma unroll
  for (int t = 0; t < kNT; ++t)
#pragma unroll
    for (int j = 0; j < 8; ++j) acc[t][j] = 0.0f;
#pragma unroll
  for (int kc = 0; kc < kKC; ++kc) {
    v16h a = afrag_ldsh(tb, lane, kc);
#pragma unroll
    for (int t = 0; t < kNT; ++t)
      acc[t] = wmma_f16(a, load_bfrag(F, lane, kc, t), acc[t]);
  }
  int half = lane >> 4, nn = lane & 15;
#pragma unroll
  for (int t = 0; t < kNT; ++t) {
    int n = t * 16 + nn;
    if (n >= kD) continue;
#pragma unroll
    for (int j = 0; j < 8; ++j) {
      int m = half * 8 + j;
      if (e0 + m < elim)
        atomicAdd(&agg[(long)shrow[warp * 16 + m] * kD + n],
                  acc[t][j] * shnorm[warp * 16 + m]);
    }
  }
}

// loop_res + combine: agg[v] = (agg[v] + rotate(x[v],loop_rel)@w_loop)/3 + bias.
__global__ __launch_bounds__(256) void k_combine(const float* __restrict__ x,
                                                 const float* __restrict__ relbuf,
                                                 const _Float16* __restrict__ F,
                                                 const float* __restrict__ bias,
                                                 float* __restrict__ agg) {
  __shared__ alignas(32) unsigned shtile[8 * kKC * 32 * 8];
  int warp = threadIdx.x >> 5, lane = threadIdx.x & 31;
  unsigned* tileu = shtile + warp * (kKC * 32 * 8);
  const _Float16* tb = (const _Float16*)tileu;
  long v0 = (long)(blockIdx.x * 8 + warp) * 16;
  const float* lr = relbuf + kRels * kD;  // loop relation row
  for (int t = lane; t < 16 * 50; t += 32) {
    int m = t / 50, k2 = 2 * (t % 50);
    long v = v0 + m;
    if (v >= kEnt) v = kEnt - 1;
    const float* xp = x + v * kDP;
    float xr0 = xp[k2], xr1 = xp[k2 + 1];
    float xi0 = xp[k2 + kDH], xi1 = xp[k2 + kDH + 1];
    float rr0 = lr[k2], rr1 = lr[k2 + 1];
    float ri0 = lr[k2 + kDH], ri1 = lr[k2 + kDH + 1];
    store_pk(tileu, m, k2, to_f16x2(xr0 * rr0 - xi0 * ri0, xr1 * rr1 - xi1 * ri1));
    store_pk(tileu, m, k2 + kDH,
             to_f16x2(xr0 * ri0 + xi0 * rr0, xr1 * ri1 + xi1 * rr1));
  }
  for (int t = lane; t < 16 * 12; t += 32)
    store_pk(tileu, t / 12, kD + 2 * (t % 12), 0u);
  __syncthreads();
  v8f acc[kNT];
#pragma unroll
  for (int t = 0; t < kNT; ++t)
#pragma unroll
    for (int j = 0; j < 8; ++j) acc[t][j] = 0.0f;
#pragma unroll
  for (int kc = 0; kc < kKC; ++kc) {
    v16h a = afrag_ldsh(tb, lane, kc);
#pragma unroll
    for (int t = 0; t < kNT; ++t)
      acc[t] = wmma_f16(a, load_bfrag(F, lane, kc, t), acc[t]);
  }
  int half = lane >> 4, nn = lane & 15;
#pragma unroll
  for (int t = 0; t < kNT; ++t) {
    int n = t * 16 + nn;
    if (n >= kD) continue;
#pragma unroll
    for (int j = 0; j < 8; ++j) {
      long v = v0 + half * 8 + j;
      if (v < kEnt) {
        long o = v * kD + n;
        agg[o] = (agg[o] + acc[t][j]) * (1.0f / 3.0f) + bias[n];
      }
    }
  }
}

// ---------------- BatchNorm / misc ----------------

__global__ void k_bn_reduce(const float* __restrict__ pre, float* __restrict__ bnsum,
                            float* __restrict__ bnsq) {
  int c = blockIdx.x;
  __shared__ float s1[256], s2[256];
  float a = 0.0f, b = 0.0f;
  for (int v = threadIdx.x; v < kEnt; v += 256) {
    float t = pre[(long)v * kD + c];
    a += t;
    b += t * t;
  }
  s1[threadIdx.x] = a;
  s2[threadIdx.x] = b;
  __syncthreads();
  for (int s = 128; s > 0; s >>= 1) {
    if (threadIdx.x < s) {
      s1[threadIdx.x] += s1[threadIdx.x + s];
      s2[threadIdx.x] += s2[threadIdx.x + s];
    }
    __syncthreads();
  }
  if (threadIdx.x == 0) {
    bnsum[c] = s1[0];
    bnsq[c] = s2[0];
  }
}

// BN + tanh; writes padded f32 x and padded f16 x. One thread per pair.
__global__ void k_bn_apply(const float* __restrict__ pre, const float* __restrict__ bnsum,
                           const float* __restrict__ bnsq, const float* __restrict__ gamma,
                           const float* __restrict__ beta, float* __restrict__ xout,
                           unsigned* __restrict__ xhf) {
  long idx = (long)blockIdx.x * blockDim.x + threadIdx.x;
  if (idx >= (long)kEnt * kDPH) return;
  int k2 = 2 * (int)(idx % kDPH);
  long v = idx / kDPH;
  float o0 = 0.0f, o1 = 0.0f;
  if (k2 < kD) {
    float mu0 = bnsum[k2] * (1.0f / kEnt), mu1 = bnsum[k2 + 1] * (1.0f / kEnt);
    float var0 = bnsq[k2] * (1.0f / kEnt) - mu0 * mu0;
    float var1 = bnsq[k2 + 1] * (1.0f / kEnt) - mu1 * mu1;
    o0 = tanhf((pre[v * kD + k2] - mu0) * rsqrtf(var0 + 1e-5f) * gamma[k2] + beta[k2]);
    o1 = tanhf((pre[v * kD + k2 + 1] - mu1) * rsqrtf(var1 + 1e-5f) * gamma[k2 + 1] +
               beta[k2 + 1]);
  }
  xout[v * kDP + k2] = o0;
  xout[v * kDP + k2 + 1] = o1;
  xhf[idx] = to_f16x2(o0, o1);
}

// r_next = (relbuf @ w_rel)[0:1000]
__global__ void k_rel_update(const float* __restrict__ relbuf,
                             const float* __restrict__ wrel, float* __restrict__ rout) {
  int idx = blockIdx.x * blockDim.x + threadIdx.x;
  if (idx >= kRels * kD) return;
  int n = idx % kD, r = idx / kD;
  float s = 0.0f;
  for (int k = 0; k < kD; ++k) s += relbuf[r * kD + k] * wrel[k * kD + n];
  rout[idx] = s;
}

// pred = x[ent] * r[rel]; write f32 to d_out section + padded f16 scratch.
__global__ void k_pred(const int* __restrict__ ent, const int* __restrict__ rel,
                       const float* __restrict__ x, const float* __restrict__ r,
                       float* __restrict__ predOut, unsigned* __restrict__ predhf) {
  int idx = blockIdx.x * blockDim.x + threadIdx.x;
  if (idx >= kBS * kDPH) return;
  int k2 = 2 * (idx % kDPH), b = idx / kDPH;
  float v0 = 0.0f, v1 = 0.0f;
  if (k2 < kD) {
    const float* xp = x + (long)ent[b] * kDP;
    const float* rp = r + (long)rel[b] * kD;
    v0 = xp[k2] * rp[k2];
    v1 = xp[k2 + 1] * rp[k2 + 1];
    predOut[b * kD + k2] = v0;
    predOut[b * kD + k2 + 1] = v1;
  }
  predhf[idx] = to_f16x2(v0, v1);
}

// logits = pred @ x^T  (M=256, K=224 padded, N=100000). Pure f16 loads + WMMA.
__global__ __launch_bounds__(256) void k_logits(const _Float16* __restrict__ predhf,
                                                const _Float16* __restrict__ xhf,
                                                float* __restrict__ out) {
  constexpr int MT = kBS / 16;          // 16
  constexpr int NTILES = kEnt / 16;     // 6250
  constexpr int NTW = 8;
  constexpr int NGROUPS = (NTILES + NTW - 1) / NTW;  // 782
  int warp = threadIdx.x >> 5, lane = threadIdx.x & 31;
  int gw = blockIdx.x * 8 + warp;
  int mt = gw / NGROUPS, ng = gw % NGROUPS;
  if (mt >= MT) return;
  v16h afr[kKC];
  {
    const _Float16* rowp = predhf + (long)(mt * 16 + (lane & 15)) * kDP;
    int kb0 = (lane >> 4) << 4;
#pragma unroll
    for (int kc = 0; kc < kKC; ++kc) {
      v4u lo = *(const v4u*)(rowp + kc * 32 + kb0);
      v4u hi = *(const v4u*)(rowp + kc * 32 + kb0 + 8);
      v8u comb;
#pragma unroll
      for (int j = 0; j < 4; ++j) {
        comb[j] = lo[j];
        comb[4 + j] = hi[j];
      }
      afr[kc] = __builtin_bit_cast(v16h, comb);
    }
  }
  int nn = lane & 15, half = lane >> 4;
  for (int tt = 0; tt < NTW; ++tt) {
    int nt = ng * NTW + tt;
    if (nt >= NTILES) break;  // wave-uniform
    int n = nt * 16 + nn;
    const _Float16* bp = xhf + (long)n * kDP + half * 16;
    v8f acc;
#pragma unroll
    for (int j = 0; j < 8; ++j) acc[j] = 0.0f;
#pragma unroll
    for (int kc = 0; kc < kKC; ++kc) {
      v16h b = *(const v16h*)(bp + kc * 32);
      acc = wmma_f16(afr[kc], b, acc);
    }
#pragma unroll
    for (int j = 0; j < 8; ++j)
      out[(long)(mt * 16 + half * 8 + j) * kEnt + n] = acc[j];
  }
}

// ---------------- host orchestration ----------------

extern "C" void kernel_launch(void* const* d_in, const int* in_sizes, int n_in,
                              void* d_out, int out_size, void* d_ws, size_t ws_size,
                              hipStream_t stream) {
  (void)in_sizes; (void)n_in; (void)out_size; (void)ws_size;
  const int*   ent       = (const int*)d_in[0];
  const int*   relq      = (const int*)d_in[1];
  const int*   edge_idx  = (const int*)d_in[2];
  const int*   edge_type = (const int*)d_in[3];
  const int*   quals     = (const int*)d_in[4];
  const float* init_emb  = (const float*)d_in[5];
  const float* init_rel  = (const float*)d_in[6];
  const float* loop_rel  = (const float*)d_in[7];
  const float* w_in      = (const float*)d_in[8];
  const float* w_out     = (const float*)d_in[9];
  const float* w_loop    = (const float*)d_in[10];
  const float* w_rel     = (const float*)d_in[11];
  const float* w_q       = (const float*)d_in[12];
  const float* conv_bias = (const float*)d_in[13];
  const float* bn_gamma  = (const float*)d_in[14];
  const float* bn_beta   = (const float*)d_in[15];

  char* ws = (char*)d_ws;
  size_t off = 0;
  auto alloc = [&](size_t bytes) {
    void* p = ws + off;
    off += (bytes + 255) & ~(size_t)255;
    return p;
  };
  float*    xA     = (float*)alloc((size_t)kEnt * kDP * 4);
  float*    xB     = (float*)alloc((size_t)kEnt * kDP * 4);
  unsigned* xhfA   = (unsigned*)alloc((size_t)kEnt * kDP * 2);
  unsigned* xhfB   = (unsigned*)alloc((size_t)kEnt * kDP * 2);
  float*    relbuf = (float*)alloc((size_t)(kRels + 1) * kD * 4);
  float*    rtmp   = (float*)alloc((size_t)kRels * kD * 4);
  float*    qsum   = (float*)alloc((size_t)kE * kD * 4);
  float*    agg    = (float*)alloc((size_t)kEnt * kD * 4);
  float*    deg    = (float*)alloc((size_t)2 * kEnt * 4);
  float*    bnsum  = (float*)alloc((size_t)kD * 4);
  float*    bnsq   = (float*)alloc((size_t)kD * 4);
  size_t fragBytes = (size_t)kKC * kNT * 32 * 16 * 2;
  unsigned* fragQ    = (unsigned*)alloc(fragBytes);
  unsigned* fragIn   = (unsigned*)alloc(fragBytes);
  unsigned* fragOut  = (unsigned*)alloc(fragBytes);
  unsigned* fragLoop = (unsigned*)alloc(fragBytes);
  unsigned* predhf   = (unsigned*)alloc((size_t)kBS * kDP * 2);

  float* logits  = (float*)d_out;
  float* predOut = logits + (size_t)kBS * kEnt;

  // padded x0 (f32 + f16); degrees (shared by both layers)
  k_init_x<<<((long)kEnt * kDPH + 255) / 256, 256, 0, stream>>>(init_emb, xA, xhfA);
  hipMemsetAsync(deg, 0, (size_t)2 * kEnt * 4, stream);
  k_degree<<<(kE + 255) / 256, 256, 0, stream>>>(edge_idx, deg);
  k_dinv<<<(2 * kEnt + 255) / 256, 256, 0, stream>>>(deg);

  float* xcur = xA;
  float* xnext = xB;
  unsigned* xhfcur = xhfA;
  unsigned* xhfnext = xhfB;
  const float* rcur = init_rel;
  const int fragBlocks = (kKC * kNT * 32 * 8 + 255) / 256;

  for (int layer = 0; layer < 2; ++layer) {
    size_t wo = (size_t)layer * kD * kD;
    k_prep_rel<<<((kRels + 1) * kD + 255) / 256, 256, 0, stream>>>(
        rcur, loop_rel + layer * kD, relbuf);
    k_prep_wfrag<<<fragBlocks, 256, 0, stream>>>(w_q + wo, fragQ);
    k_prep_wfrag<<<fragBlocks, 256, 0, stream>>>(w_in + wo, fragIn);
    k_prep_wfrag<<<fragBlocks, 256, 0, stream>>>(w_out + wo, fragOut);
    k_prep_wfrag<<<fragBlocks, 256, 0, stream>>>(w_loop + wo, fragLoop);
    hipMemsetAsync(qsum, 0, (size_t)kE * kD * 4, stream);
    hipMemsetAsync(agg, 0, (size_t)kEnt * kD * 4, stream);

    k_qual<<<((long)kNQ * kDH + 255) / 256, 256, 0, stream>>>(quals, xcur, relbuf,
                                                              qsum);
    k_gemm_inplace<<<(kE + 127) / 128, 256, 0, stream>>>(
        qsum, (const _Float16*)fragQ, (long)kE);

    k_message<<<(kHalfE + 127) / 128, 256, 0, stream>>>(
        edge_idx, edge_type, xcur, relbuf, qsum, deg, (const _Float16*)fragIn, agg,
        0);
    k_message<<<(kHalfE + 127) / 128, 256, 0, stream>>>(
        edge_idx, edge_type, xcur, relbuf, qsum, deg + kEnt,
        (const _Float16*)fragOut, agg, kHalfE);

    k_combine<<<(kEnt + 127) / 128, 256, 0, stream>>>(
        xcur, relbuf, (const _Float16*)fragLoop, conv_bias + layer * kD, agg);
    k_bn_reduce<<<kD, 256, 0, stream>>>(agg, bnsum, bnsq);
    k_bn_apply<<<((long)kEnt * kDPH + 255) / 256, 256, 0, stream>>>(
        agg, bnsum, bnsq, bn_gamma + layer * kD, bn_beta + layer * kD, xnext,
        xhfnext);
    k_rel_update<<<(kRels * kD + 255) / 256, 256, 0, stream>>>(relbuf, w_rel + wo,
                                                               rtmp);
    rcur = rtmp;
    float* t = xcur; xcur = xnext; xnext = t;
    unsigned* th = xhfcur; xhfcur = xhfnext; xhfnext = th;
  }

  // decoder
  k_pred<<<(kBS * kDPH + 255) / 256, 256, 0, stream>>>(ent, relq, xcur, rtmp,
                                                       predOut, predhf);
  constexpr int NGROUPS = (kEnt / 16 + 7) / 8;  // 782
  k_logits<<<(16 * NGROUPS + 7) / 8, 256, 0, stream>>>(
      (const _Float16*)predhf, (const _Float16*)xhfcur, logits);
}
